// SentenceGeneration_60078002537026
// MI455X (gfx1250) — compile-verified, matrix operand order checked by reading, not run on hardware
//
#include <hip/hip_runtime.h>

// Problem sizes (fixed by reference)
#define VV    50257
#define EE    1024
#define HH    1024
#define BB    64
#define TT    512
#define NPAD  50272      // VV padded up to multiple of 16 for WMMA N-tiles

typedef __attribute__((ext_vector_type(16))) __bf16 v16bf;
typedef __attribute__((ext_vector_type(8)))  float  v8f;
typedef __attribute__((ext_vector_type(4)))  float  v4f;

union FragBF { v16bf v; v4f q[2]; };
union BF8    { __bf16 h[8]; v4f q; };

// ---------------------------------------------------------------------------
// WMMA helpers (CDNA5: v_wmma_f32_16x16x32_bf16, wave32, fp32 accumulate)
// ---------------------------------------------------------------------------
__device__ __forceinline__ v8f wmma_bf16(v16bf a, v16bf b, v8f c) {
    // (neg_a, A, neg_b, B, c_mod, C, reuse_a, reuse_b)
    return __builtin_amdgcn_wmma_f32_16x16x32_bf16(false, a, false, b, (short)0, c,
                                                   false, false);
}

// A/B fragment load for row-major [rows][stride] bf16 data.
// ISA 16-bit 16x32 layout: lane l -> row (l&15); halves (l>>4)*8..+7 in v0-3,
// +16.. in v4-7.  Two 16-byte loads per lane.
__device__ __forceinline__ v16bf load_tile_bf16(const __bf16* __restrict__ base,
                                                int stride, int row0, int k0,
                                                int lane) {
    const int r  = lane & 15;
    const int hs = lane >> 4;
    const __bf16* p = base + (size_t)(row0 + r) * (size_t)stride
                           + (size_t)(k0 + hs * 8);
    FragBF f;
    f.q[0] = *reinterpret_cast<const v4f*>(p);
    f.q[1] = *reinterpret_cast<const v4f*>(p + 16);
    return f.v;
}

__device__ __forceinline__ float sig_(float x) {
    return 1.0f / (1.0f + __expf(-x));
}

// ---------------------------------------------------------------------------
// 1) fp32 -> bf16 conversion (8 elems / thread, 128-bit loads & stores)
// ---------------------------------------------------------------------------
__global__ void cvt_f32_bf16_kernel(const float* __restrict__ src,
                                    __bf16* __restrict__ dst, long long n8) {
    long long i = (long long)blockIdx.x * blockDim.x + threadIdx.x;
    if (i >= n8) return;
    v4f a = reinterpret_cast<const v4f*>(src)[2 * i];
    v4f b = reinterpret_cast<const v4f*>(src)[2 * i + 1];
    BF8 o;
    o.h[0] = (__bf16)a[0]; o.h[1] = (__bf16)a[1];
    o.h[2] = (__bf16)a[2]; o.h[3] = (__bf16)a[3];
    o.h[4] = (__bf16)b[0]; o.h[5] = (__bf16)b[1];
    o.h[6] = (__bf16)b[2]; o.h[7] = (__bf16)b[3];
    reinterpret_cast<v4f*>(dst)[i] = o.q;
}

__global__ void fill_zero_bf16_kernel(__bf16* __restrict__ dst, long long n8) {
    long long i = (long long)blockIdx.x * blockDim.x + threadIdx.x;
    if (i >= n8) return;
    v4f z = {0.f, 0.f, 0.f, 0.f};
    reinterpret_cast<v4f*>(dst)[i] = z;
}

// ---------------------------------------------------------------------------
// 2) Embedding gather + cast, time-major output x[T][B][E] (bf16)
// ---------------------------------------------------------------------------
__global__ void embed_gather_kernel(const int* __restrict__ tokens,
                                    const float* __restrict__ emb,
                                    __bf16* __restrict__ xbf) {
    const int EP8 = EE / 8;
    long long i = (long long)blockIdx.x * blockDim.x + threadIdx.x;
    if (i >= (long long)TT * BB * EP8) return;
    int e8 = (int)(i % EP8);
    long long tb = i / EP8;
    int b = (int)(tb % BB);
    int t = (int)(tb / BB);
    int tok = tokens[(long long)b * TT + t];        // batch_reviews[b][t]
    const v4f* s = reinterpret_cast<const v4f*>(emb + (size_t)tok * EE + e8 * 8);
    v4f a = s[0], c = s[1];
    BF8 o;
    o.h[0] = (__bf16)a[0]; o.h[1] = (__bf16)a[1];
    o.h[2] = (__bf16)a[2]; o.h[3] = (__bf16)a[3];
    o.h[4] = (__bf16)c[0]; o.h[5] = (__bf16)c[1];
    o.h[6] = (__bf16)c[2]; o.h[7] = (__bf16)c[3];
    __bf16* d = xbf + ((size_t)t * BB + b) * EE + (size_t)e8 * 8;
    *reinterpret_cast<v4f*>(d) = o.q;
}

// ---------------------------------------------------------------------------
// 3) Zero initial state
// ---------------------------------------------------------------------------
__global__ void init_state_kernel(__bf16* __restrict__ hb0,
                                  float* __restrict__ c) {
    int i = blockIdx.x * blockDim.x + threadIdx.x;
    if (i < BB * HH) {
        hb0[i] = (__bf16)0.0f;
        c[i]   = 0.0f;
    }
}

// ---------------------------------------------------------------------------
// 4) One LSTM step.  64 waves; wave jt owns hidden cols [16*jt, 16*jt+16).
//    Each wave computes gates i,f,g,o for those cols over all 64 batch rows
//    (16 f32 accumulators = 4 gates x 4 M-tiles), then does the cell update.
//    gates = x_t @ W_ih^T + h @ W_hh^T  (K = 1024 + 1024, bf16 WMMA k=32)
// ---------------------------------------------------------------------------
__global__ void __launch_bounds__(128)
lstm_step_kernel(const __bf16* __restrict__ xbf,
                 const __bf16* __restrict__ Wih,   // [4H][E] bf16
                 const __bf16* __restrict__ Whh,   // [4H][H] bf16
                 const __bf16* __restrict__ hin,   // [B][H] bf16 (t-1)
                 __bf16* __restrict__ hout,        // [B][H] bf16 (t)
                 float* __restrict__ hf,           // [B][H] f32 (for output)
                 float* __restrict__ c,            // [B][H] f32 (in/out)
                 int t) {
    const int lane = threadIdx.x & 31;
    const int jt   = blockIdx.x * 4 + (threadIdx.x >> 5);   // 0..63
    const int j0   = jt * 16;

    v8f acc[4][4];
#pragma unroll
    for (int g = 0; g < 4; ++g)
#pragma unroll
        for (int mt = 0; mt < 4; ++mt)
            acc[g][mt] = (v8f){0.f, 0.f, 0.f, 0.f, 0.f, 0.f, 0.f, 0.f};

    // Phase 0: x_t contribution (A = x_t [64][1024], B rows = W_ih[g*H + j])
    const __bf16* A0 = xbf + (size_t)t * (BB * EE);
#pragma unroll 2
    for (int kt = 0; kt < EE / 32; ++kt) {
        const int k0 = kt * 32;
        v16bf a[4];
#pragma unroll
        for (int mt = 0; mt < 4; ++mt)
            a[mt] = load_tile_bf16(A0, EE, mt * 16, k0, lane);
#pragma unroll
        for (int g = 0; g < 4; ++g) {
            v16bf bm = load_tile_bf16(Wih, EE, g * HH + j0, k0, lane);
#pragma unroll
            for (int mt = 0; mt < 4; ++mt)
                acc[g][mt] = wmma_bf16(a[mt], bm, acc[g][mt]);
        }
    }

    // Phase 1: recurrent contribution (A = h_{t-1}, B rows = W_hh[g*H + j])
#pragma unroll 2
    for (int kt = 0; kt < HH / 32; ++kt) {
        const int k0 = kt * 32;
        v16bf a[4];
#pragma unroll
        for (int mt = 0; mt < 4; ++mt)
            a[mt] = load_tile_bf16(hin, HH, mt * 16, k0, lane);
#pragma unroll
        for (int g = 0; g < 4; ++g) {
            v16bf bm = load_tile_bf16(Whh, HH, g * HH + j0, k0, lane);
#pragma unroll
            for (int mt = 0; mt < 4; ++mt)
                acc[g][mt] = wmma_bf16(a[mt], bm, acc[g][mt]);
        }
    }

    // Cell update.  C/D layout: vgpr r, lane l -> m = r + 8*(l>>4), n = l&15.
    const int n  = j0 + (lane & 15);
    const int hs = lane >> 4;
#pragma unroll
    for (int mt = 0; mt < 4; ++mt) {
#pragma unroll
        for (int r = 0; r < 8; ++r) {
            const int m = mt * 16 + r + hs * 8;
            const size_t idx = (size_t)m * HH + n;
            const float iv = acc[0][mt][r];
            const float fv = acc[1][mt][r];
            const float gv = acc[2][mt][r];
            const float ov = acc[3][mt][r];
            const float c2 = sig_(fv) * c[idx] + sig_(iv) * tanhf(gv);
            const float h2 = sig_(ov) * tanhf(c2);
            c[idx]    = c2;
            hf[idx]   = h2;
            hout[idx] = (__bf16)h2;
        }
    }
}

// ---------------------------------------------------------------------------
// 5) Classifier: logits[64][V] = h @ W_cls^T + b_cls.
//    One wave per 16-wide vocab tile, 4 M-tiles, K=1024 in 32 WMMA steps.
// ---------------------------------------------------------------------------
__global__ void __launch_bounds__(256)
classifier_kernel(const __bf16* __restrict__ hb,     // [B][H] bf16 (final h)
                  const __bf16* __restrict__ Wc,     // [NPAD][H] bf16
                  const float* __restrict__ bcls,    // [V]
                  float* __restrict__ out) {         // logits at out[0..]
    const int lane = threadIdx.x & 31;
    const int nt   = blockIdx.x * 8 + (threadIdx.x >> 5);
    if (nt >= NPAD / 16) return;
    const int n0 = nt * 16;

    v8f acc[4];
#pragma unroll
    for (int mt = 0; mt < 4; ++mt)
        acc[mt] = (v8f){0.f, 0.f, 0.f, 0.f, 0.f, 0.f, 0.f, 0.f};

#pragma unroll 2
    for (int kt = 0; kt < HH / 32; ++kt) {
        const int k0 = kt * 32;
        if (kt + 1 < HH / 32)   // pull next W tile toward L2/L0
            __builtin_prefetch(Wc + (size_t)(n0 + (lane & 15)) * HH + k0 + 32, 0, 0);
        v16bf bm = load_tile_bf16(Wc, HH, n0, k0, lane);
#pragma unroll
        for (int mt = 0; mt < 4; ++mt) {
            v16bf a = load_tile_bf16(hb, HH, mt * 16, k0, lane);
            acc[mt] = wmma_bf16(a, bm, acc[mt]);
        }
    }

    const int n  = n0 + (lane & 15);
    const int hs = lane >> 4;
    if (n < VV) {
        const float bias = bcls[n];
#pragma unroll
        for (int mt = 0; mt < 4; ++mt)
#pragma unroll
            for (int r = 0; r < 8; ++r) {
                const int m = mt * 16 + r + hs * 8;
                out[(size_t)m * VV + n] = acc[mt][r] + bias;
            }
    }
}

// ---------------------------------------------------------------------------
// 6) Append final h and c (fp32) after logits in d_out
// ---------------------------------------------------------------------------
__global__ void write_state_kernel(const float* __restrict__ hf,
                                   const float* __restrict__ c,
                                   float* __restrict__ out) {
    int i = blockIdx.x * blockDim.x + threadIdx.x;
    if (i < BB * HH) {
        out[(size_t)BB * VV + i]           = hf[i];
        out[(size_t)BB * VV + BB * HH + i] = c[i];
    }
}

// ---------------------------------------------------------------------------
// Host side
// ---------------------------------------------------------------------------
extern "C" void kernel_launch(void* const* d_in, const int* in_sizes, int n_in,
                              void* d_out, int out_size, void* d_ws, size_t ws_size,
                              hipStream_t stream) {
    const int*   tokens = (const int*)  d_in[0];   // [B][T]
    const float* emb    = (const float*)d_in[1];   // [V][E]
    const float* W_ih   = (const float*)d_in[2];   // [4H][E]
    const float* W_hh   = (const float*)d_in[3];   // [4H][H]
    const float* W_cls  = (const float*)d_in[4];   // [V][H]
    const float* b_cls  = (const float*)d_in[5];   // [V]
    float* out = (float*)d_out;

    // Workspace layout (256B aligned)
    char* ws = (char*)d_ws;
    size_t off = 0;
    auto alloc = [&](size_t bytes) {
        void* p = ws + off;
        off = (off + bytes + 255) & ~(size_t)255;
        return p;
    };
    __bf16* xbf     = (__bf16*)alloc((size_t)TT * BB * EE * 2);   //  64 MB
    __bf16* Wih_bf  = (__bf16*)alloc((size_t)4 * HH * EE * 2);    //   8 MB
    __bf16* Whh_bf  = (__bf16*)alloc((size_t)4 * HH * HH * 2);    //   8 MB
    __bf16* Wcls_bf = (__bf16*)alloc((size_t)NPAD * HH * 2);      // ~98 MB
    __bf16* hb0     = (__bf16*)alloc((size_t)BB * HH * 2);
    __bf16* hb1     = (__bf16*)alloc((size_t)BB * HH * 2);
    float*  hf      = (float*) alloc((size_t)BB * HH * 4);
    float*  cst     = (float*) alloc((size_t)BB * HH * 4);

    const int THR = 256;
    auto blocks8 = [](long long n8) { return (int)((n8 + 255) / 256); };

    // Weight conversion fp32 -> bf16
    long long n8;
    n8 = (long long)4 * HH * EE / 8;
    cvt_f32_bf16_kernel<<<blocks8(n8), THR, 0, stream>>>(W_ih, Wih_bf, n8);
    n8 = (long long)4 * HH * HH / 8;
    cvt_f32_bf16_kernel<<<blocks8(n8), THR, 0, stream>>>(W_hh, Whh_bf, n8);
    n8 = (long long)VV * HH / 8;
    cvt_f32_bf16_kernel<<<blocks8(n8), THR, 0, stream>>>(W_cls, Wcls_bf, n8);
    // Zero the vocab padding rows [VV, NPAD)
    n8 = (long long)(NPAD - VV) * HH / 8;
    fill_zero_bf16_kernel<<<blocks8(n8), THR, 0, stream>>>(
        Wcls_bf + (size_t)VV * HH, n8);

    // Embedding gather -> bf16 time-major x[T][B][E]
    long long nemb = (long long)TT * BB * (EE / 8);
    embed_gather_kernel<<<blocks8(nemb), THR, 0, stream>>>(tokens, emb, xbf);

    // Zero initial h, c
    init_state_kernel<<<(BB * HH + THR - 1) / THR, THR, 0, stream>>>(hb0, cst);

    // Sequential LSTM scan: one launch per step, h double-buffered
    for (int t = 0; t < TT; ++t) {
        const __bf16* hin = (t & 1) ? hb1 : hb0;
        __bf16* hout      = (t & 1) ? hb0 : hb1;
        lstm_step_kernel<<<16, 128, 0, stream>>>(xbf, Wih_bf, Whh_bf,
                                                 hin, hout, hf, cst, t);
    }
    // TT = 512 (even): last step (t=511, odd) wrote hb0 -> final h is hb0
    const __bf16* hfinal = hb0;

    // Classifier GEMM -> logits
    const int n_tiles = NPAD / 16;               // 3142
    classifier_kernel<<<(n_tiles + 7) / 8, 256, 0, stream>>>(hfinal, Wcls_bf,
                                                             b_cls, out);

    // Append final h, c
    write_state_kernel<<<(BB * HH + THR - 1) / THR, THR, 0, stream>>>(hf, cst, out);

    (void)in_sizes; (void)n_in; (void)out_size; (void)ws_size;
}
